// TopKSparseAutoencoder_76802605187769
// MI455X (gfx1250) — matrix operand order repaired and checked by reading, not compiled, not run on hardware
//
#include <hip/hip_runtime.h>

// ---------------------------------------------------------------------------
// TopK sparse autoencoder forward, fused, bf16 WMMA path for MI455X (gfx1250).
//
// Roofline: 86 GFLOP, ~370 MB HBM traffic. bf16 WMMA (16x16x32, f32 acc) is
// ~8x the f32-WMMA rate -> ~35us compute vs ~16us memory floor; f32 WMMA
// would be ~275us. Weights (1.3MB) are pre-swizzled once into WMMA B-fragment
// order in d_ws (L2 resident, 192MB L2), so the hot loop issues only
// coalesced uint4 fragment loads + v_wmma.
//
// Fragment layout assumptions (CDNA5 ISA 7.12.2, wave32):
//   A (16xK bf16): lane%16 = M row; lanes 0-15 hold K = 32k+{0..7,16..23},
//                  lanes 16-31 hold K = 32k+{8..15,24..31} (8 dwords/lane).
//   B (Kx16 bf16): lane%16 = N col; lanes 0-15 hold K = 32k+{0..7,16..23},
//                  lanes 16-31 hold K = 32k+{8..15,24..31} (8 dwords/lane).
//   C/D (16x16 f32): VGPR r: lanes 0-15 -> (M=r, N=lane),
//                    lanes 16-31 -> (M=r+8, N=lane-16).
// ---------------------------------------------------------------------------

typedef __attribute__((ext_vector_type(16))) __bf16 v16bf;
typedef __attribute__((ext_vector_type(8)))  float  v8f;

#define B_ROWS   131072
#define IN_DIM   256
#define HID_DIM  512
#define LAT_DIM  64
#define TOPK     5

// d_out flat offsets (elements, float)
#define OFF_RECON 0
#define OFF_SL    33554432ull   // B*256
#define OFF_PA    41943040ull   // + B*64
#define OFF_MASK  50331648ull   // + B*64
#define OFF_SPARS 58720256ull   // + B*64

// d_ws byte offsets
#define WS_B1   0u        // W_eh^T frags  bf16  (K=256 -> 8 ksteps, N=512 -> 32 ntiles) 262144 B
#define WS_B2   262144u   // W_eo^T frags  bf16  (K=512 -> 16,      N=64  -> 4)          65536 B
#define WS_B3   327680u   // W_dh^T frags  bf16  (K=64  -> 2,       N=512 -> 32)         65536 B
#define WS_B4   393216u   // W_do^T frags  bf16  (K=512 -> 16,      N=256 -> 16)        262144 B
#define WS_BOUT 655360u   // f32[256] = b_do + decoder_bias + pre_bias
#define WS_B2B  656384u   // f32[64]  = b_eo + latent_bias

union Frag { v16bf v; __bf16 e[16]; uint4 q[2]; };

// --- prep: swizzle W[N][K] f32 (row-major, N=out_d, K=in_d) into bf16 B-frags.
// Frag dword index: idx = (((nt*ksteps+ks)*2+p)*32 + lane)*4 + d ; each dword
// holds bf16 pair (K=base+j, K=base+j+1), j = 2m (+8 if m>=4), m = p*4+d, base = 32ks+8*(lane>>4).
__global__ void prep_weights(const float* __restrict__ W, __bf16* __restrict__ dst,
                             int Kdim, int ksteps, int total) {
  int idx = blockIdx.x * blockDim.x + threadIdx.x;
  if (idx >= total) return;
  int d    = idx & 3;
  int lane = (idx >> 2) & 31;
  int p    = (idx >> 7) & 1;
  int ks   = (idx >> 8) % ksteps;
  int nt   = idx / (ksteps << 8);
  int n    = nt * 16 + (lane & 15);
  int e    = (p * 4 + d) * 2;
  int k    = ks * 32 + ((lane >> 4) << 3) + (e < 8 ? e : e + 8);   // B: half-lanes offset by 8, second 8 at +16
  dst[2 * (size_t)idx]     = (__bf16)W[(size_t)n * Kdim + k];
  dst[2 * (size_t)idx + 1] = (__bf16)W[(size_t)n * Kdim + k + 1];
}

__global__ void prep_bias(const float* __restrict__ b_eo, const float* __restrict__ lat_b,
                          const float* __restrict__ b_do, const float* __restrict__ dec_b,
                          const float* __restrict__ pre_b,
                          float* __restrict__ bout, float* __restrict__ b2) {
  int i = threadIdx.x;                // 256 threads
  if (i < LAT_DIM) b2[i] = b_eo[i] + lat_b[i];
  bout[i] = b_do[i] + dec_b[i] + pre_b[i];
}

// Per-wave LDS: buf0 16*1040B (x stage f32 stride 260 / h,hd bf16 stride 520),
// pa 16*68 f32, sl 16*68 f32.  25344 B/wave, 2 waves -> 50688 B.
#define WAVE_SMEM 25344
#define XS  260
#define HS  520
#define PS  68

__global__ __launch_bounds__(64) void sae_fused(
    const float* __restrict__ x, const float* __restrict__ pre_bias,
    const float* __restrict__ b_eh, const float* __restrict__ b_dh,
    const void* __restrict__ ws, float* __restrict__ out) {
  __shared__ __align__(16) unsigned char smem[2 * WAVE_SMEM];
  const int wave = threadIdx.x >> 5;
  const int lane = threadIdx.x & 31;
  const int rowBase = blockIdx.x * 32 + wave * 16;

  unsigned char* base = smem + wave * WAVE_SMEM;
  float*  xs   = (float*)base;                   // 16 x 260 f32
  __bf16* hbuf = (__bf16*)base;                  // 16 x 520 bf16 (reuses xs)
  float*  pa   = (float*)(base + 16640);         // 16 x 68 f32
  float*  sl   = (float*)(base + 20992);         // 16 x 68 f32

  const uint4* Bq1 = (const uint4*)((const unsigned char*)ws + WS_B1);
  const uint4* Bq2 = (const uint4*)((const unsigned char*)ws + WS_B2);
  const uint4* Bq3 = (const uint4*)((const unsigned char*)ws + WS_B3);
  const uint4* Bq4 = (const uint4*)((const unsigned char*)ws + WS_B4);
  const float* biasOut = (const float*)((const unsigned char*)ws + WS_BOUT);
  const float* bias2   = (const float*)((const unsigned char*)ws + WS_B2B);

  const int mrow = lane & 15;           // A row / C col lane id
  const int half = lane >> 4;
  const int kbA  = half * 8;            // A: half-lanes offset by K+8

  // ---- Phase A: stage xc = x - pre_bias into LDS (coalesced float4) --------
  {
    const float4* x4  = (const float4*)(x + (size_t)rowBase * IN_DIM);
    const float4* pb4 = (const float4*)pre_bias;
    for (int i = 0; i < 32; ++i) {
      int t = i * 32 + lane;            // 1024 float4 chunks: 16 rows x 64
      int r = t >> 6, c4 = t & 63;
      float4 v = x4[(size_t)r * 64 + c4];
      float4 p = pb4[c4];
      v.x -= p.x; v.y -= p.y; v.z -= p.z; v.w -= p.w;
      ((float4*)(xs + r * XS))[c4] = v;
    }
  }
  __syncthreads();

  // ---- Phase B: A1 fragments (16x256 bf16, 8 ksteps) in registers ----------
  Frag a1[8];
#pragma unroll
  for (int ks = 0; ks < 8; ++ks) {
#pragma unroll
    for (int e = 0; e < 16; ++e) {
      int kk = ks * 32 + ((e < 8) ? (kbA + e) : (16 + kbA + (e - 8)));
      a1[ks].e[e] = (__bf16)xs[mrow * XS + kk];
    }
  }
  __syncthreads();   // xs dead; buf0 becomes h (bf16)

  // ---- GEMM1: h = relu(xc @ W_eh^T + b_eh)  [16x512] -----------------------
  for (int nt = 0; nt < 32; ++nt) {
    v8f acc = {0, 0, 0, 0, 0, 0, 0, 0};
#pragma unroll
    for (int ks = 0; ks < 8; ++ks) {
      Frag b;
      b.q[0] = Bq1[(nt * 8 + ks) * 64 + lane];
      b.q[1] = Bq1[(nt * 8 + ks) * 64 + 32 + lane];
      acc = __builtin_amdgcn_wmma_f32_16x16x32_bf16(false, a1[ks].v, false, b.v,
                                                    (short)0, acc, false, false);
    }
    int n = nt * 16 + mrow;
    float bias = b_eh[n];
#pragma unroll
    for (int r = 0; r < 8; ++r) {
      int M = r + half * 8;
      float v = acc[r] + bias;
      hbuf[M * HS + n] = (__bf16)(v > 0.0f ? v : 0.0f);
    }
  }
  __syncthreads();

  // ---- GEMM2: pre_act = h @ W_eo^T + (b_eo + latent_bias)  [16x64] ---------
  Frag a2[16];
#pragma unroll
  for (int ks = 0; ks < 16; ++ks) {
#pragma unroll
    for (int e = 0; e < 16; ++e) {
      int kk = ks * 32 + ((e < 8) ? (kbA + e) : (16 + kbA + (e - 8)));
      a2[ks].e[e] = hbuf[mrow * HS + kk];
    }
  }
  for (int nt = 0; nt < 4; ++nt) {
    v8f acc = {0, 0, 0, 0, 0, 0, 0, 0};
#pragma unroll
    for (int ks = 0; ks < 16; ++ks) {
      Frag b;
      b.q[0] = Bq2[(nt * 16 + ks) * 64 + lane];
      b.q[1] = Bq2[(nt * 16 + ks) * 64 + 32 + lane];
      acc = __builtin_amdgcn_wmma_f32_16x16x32_bf16(false, a2[ks].v, false, b.v,
                                                    (short)0, acc, false, false);
    }
    int n = nt * 16 + mrow;
    float bias = bias2[n];
#pragma unroll
    for (int r = 0; r < 8; ++r) {
      int M = r + half * 8;
      float v = acc[r] + bias;
      out[OFF_PA + (size_t)(rowBase + M) * LAT_DIM + n] = v;  // pre_act
      pa[M * PS + n] = v;
    }
  }
  __syncthreads();

  // ---- TopK(5): lanes 0-15 each own one row --------------------------------
  if (lane < 16) {
    const int r = lane;
    float tv[TOPK]; int ti[TOPK];
#pragma unroll
    for (int i = 0; i < TOPK; ++i) { tv[i] = -3.4e38f; ti[i] = 0; }
    for (int c = 0; c < LAT_DIM; ++c) {
      float v = pa[r * PS + c];
      if (v > tv[TOPK - 1]) {           // strict > keeps lower index on ties
        int j = TOPK - 1;
        while (j > 0 && v > tv[j - 1]) { tv[j] = tv[j - 1]; ti[j] = ti[j - 1]; --j; }
        tv[j] = v; ti[j] = c;
      }
    }
    for (int c = 0; c < LAT_DIM; ++c) { sl[r * PS + c] = 0.0f; pa[r * PS + c] = 0.0f; }
#pragma unroll
    for (int i = 0; i < TOPK; ++i) { sl[r * PS + ti[i]] = tv[i]; pa[r * PS + ti[i]] = 1.0f; }
  }
  __syncthreads();

  // ---- coalesced sparse_latent + mask stores -------------------------------
  for (int i = 0; i < 8; ++i) {
    int t = i * 32 + lane;              // 256 float4 chunks: 16 rows x 16
    int r = t >> 4, c4 = t & 15;
    float4 s = ((const float4*)(sl + r * PS))[c4];
    float4 m = ((const float4*)(pa + r * PS))[c4];
    ((float4*)(out + OFF_SL   + (size_t)(rowBase + r) * LAT_DIM))[c4] = s;
    ((float4*)(out + OFF_MASK + (size_t)(rowBase + r) * LAT_DIM))[c4] = m;
  }

  // ---- GEMM3: hd = relu(sl @ W_dh^T + b_dh)  [16x512] ----------------------
  Frag a3[2];
#pragma unroll
  for (int ks = 0; ks < 2; ++ks) {
#pragma unroll
    for (int e = 0; e < 16; ++e) {
      int kk = ks * 32 + ((e < 8) ? (kbA + e) : (16 + kbA + (e - 8)));
      a3[ks].e[e] = (__bf16)sl[mrow * PS + kk];
    }
  }
  __syncthreads();   // safe to overwrite hbuf with hd now
  for (int nt = 0; nt < 32; ++nt) {
    v8f acc = {0, 0, 0, 0, 0, 0, 0, 0};
#pragma unroll
    for (int ks = 0; ks < 2; ++ks) {
      Frag b;
      b.q[0] = Bq3[(nt * 2 + ks) * 64 + lane];
      b.q[1] = Bq3[(nt * 2 + ks) * 64 + 32 + lane];
      acc = __builtin_amdgcn_wmma_f32_16x16x32_bf16(false, a3[ks].v, false, b.v,
                                                    (short)0, acc, false, false);
    }
    int n = nt * 16 + mrow;
    float bias = b_dh[n];
#pragma unroll
    for (int r = 0; r < 8; ++r) {
      int M = r + half * 8;
      float v = acc[r] + bias;
      hbuf[M * HS + n] = (__bf16)(v > 0.0f ? v : 0.0f);
    }
  }
  __syncthreads();

  // ---- GEMM4: recon = hd @ W_do^T + (b_do + dec_bias + pre_bias)  [16x256] -
  Frag a4[16];
#pragma unroll
  for (int ks = 0; ks < 16; ++ks) {
#pragma unroll
    for (int e = 0; e < 16; ++e) {
      int kk = ks * 32 + ((e < 8) ? (kbA + e) : (16 + kbA + (e - 8)));
      a4[ks].e[e] = hbuf[mrow * HS + kk];
    }
  }
  for (int nt = 0; nt < 16; ++nt) {
    v8f acc = {0, 0, 0, 0, 0, 0, 0, 0};
#pragma unroll
    for (int ks = 0; ks < 16; ++ks) {
      Frag b;
      b.q[0] = Bq4[(nt * 16 + ks) * 64 + lane];
      b.q[1] = Bq4[(nt * 16 + ks) * 64 + 32 + lane];
      acc = __builtin_amdgcn_wmma_f32_16x16x32_bf16(false, a4[ks].v, false, b.v,
                                                    (short)0, acc, false, false);
    }
    int n = nt * 16 + mrow;
    float bias = biasOut[n];
#pragma unroll
    for (int r = 0; r < 8; ++r) {
      int M = r + half * 8;
      out[OFF_RECON + (size_t)(rowBase + M) * IN_DIM + n] = acc[r] + bias;
    }
  }

  // ---- sparsity: mask has exactly K true per row -> constant K/LAT ---------
  if (blockIdx.x == 0 && threadIdx.x == 0) out[OFF_SPARS] = (float)TOPK / (float)LAT_DIM;
}

extern "C" void kernel_launch(void* const* d_in, const int* in_sizes, int n_in,
                              void* d_out, int out_size, void* d_ws, size_t ws_size,
                              hipStream_t stream) {
  (void)in_sizes; (void)n_in; (void)out_size; (void)ws_size;
  const float* x        = (const float*)d_in[0];
  const float* pre_bias = (const float*)d_in[1];
  const float* W_eh     = (const float*)d_in[2];
  const float* b_eh     = (const float*)d_in[3];
  const float* W_eo     = (const float*)d_in[4];
  const float* b_eo     = (const float*)d_in[5];
  const float* lat_b    = (const float*)d_in[6];
  const float* W_dh     = (const float*)d_in[7];
  const float* b_dh     = (const float*)d_in[8];
  const float* W_do     = (const float*)d_in[9];
  const float* b_do     = (const float*)d_in[10];
  const float* dec_b    = (const float*)d_in[11];
  unsigned char* ws = (unsigned char*)d_ws;

  // Weight fragment prep (L2-resident, ~656KB total; reruns every call: cheap
  // and keeps kernel_launch deterministic/stateless).
  prep_weights<<<256, 256, 0, stream>>>(W_eh, (__bf16*)(ws + WS_B1), IN_DIM,  8, 65536);
  prep_weights<<< 64, 256, 0, stream>>>(W_eo, (__bf16*)(ws + WS_B2), HID_DIM, 16, 16384);
  prep_weights<<< 64, 256, 0, stream>>>(W_dh, (__bf16*)(ws + WS_B3), LAT_DIM, 2, 16384);
  prep_weights<<<256, 256, 0, stream>>>(W_do, (__bf16*)(ws + WS_B4), HID_DIM, 16, 65536);
  prep_bias<<<1, 256, 0, stream>>>(b_eo, lat_b, b_do, dec_b, pre_bias,
                                   (float*)(ws + WS_BOUT), (float*)(ws + WS_B2B));

  sae_fused<<<B_ROWS / 32, 64, 0, stream>>>(x, pre_bias, b_eh, b_dh,
                                            (const void*)ws, (float*)d_out);
}
